// QueryEncoder_82660940579266
// MI455X (gfx1250) — matrix operand order repaired and checked by reading, not compiled, not run on hardware
//
#include <hip/hip_runtime.h>

// ---- CDNA5 (gfx1250) types ----
typedef __bf16 bf16;
typedef __bf16 v16bf __attribute__((ext_vector_type(16)));
typedef float  v8f   __attribute__((ext_vector_type(8)));
typedef float  v4f   __attribute__((ext_vector_type(4)));
typedef int    v4i_  __attribute__((vector_size(16)));   // matches builtin param type

union Frag32 { v4f f4[2]; v16bf v; };  // 32B: one WMMA bf16 A/B fragment per lane

__device__ __forceinline__ float sigf(float x) { return 1.f / (1.f + __expf(-x)); }

// gfx1250 async global->LDS path (guarded: falls back to reg staging if absent)
#if defined(__HIP_DEVICE_COMPILE__) && __has_builtin(__builtin_amdgcn_global_load_async_to_lds_b128)
#define USE_ASYNC 1
#else
#define USE_ASYNC 0
#endif

template<int N> __device__ __forceinline__ void wait_async() {
#if defined(__HIP_DEVICE_COMPILE__)
#if __has_builtin(__builtin_amdgcn_s_wait_asynccnt)
    __builtin_amdgcn_s_wait_asynccnt(N);
#else
    asm volatile("s_wait_asynccnt %0" :: "n"(N));
#endif
#endif
}

// Problem constants (reference: FEW=128, BATCH=4096, D=512, STEPS=4, H=2D=1024)
#define NB    4096          // batch
#define DIM   512
#define NFEW  128
#define KCAT  1536          // D + 2D
#define NH    1024          // hidden 2D

// =====================================================================
// Generic NT GEMM: C(MxN) = A(MxK) * B(NxK)^T, bf16 in, f32 acc.
// Block tile 128x128, K-step 32, 256 threads = 8 wave32 (2 M x 4 N waves),
// each wave: 4x2 tiles of v_wmma_f32_16x16x32_bf16.
// Staging: triple-buffered global_load_async_to_lds_b128 (ASYNCcnt) when
// available, else double-buffered VGPR staging.
// MODE 0: gates kernel. N-cols map to 4 gate quadrants of a 32-wide j
//         block; epilogue does the fused LSTMCell update.
// MODE 1: store f32 C (attention logits).
// MODE 2: store bf16 C (r written into xcat's r slot).
// =====================================================================
template<int MODE>
__global__ void __launch_bounds__(256)
gemm_nt(const bf16* __restrict__ A, int lda,
        const bf16* __restrict__ B, int ldb, int K,
        const float* __restrict__ bsum,   // MODE0: b_ih+b_hh  (4096)
        float* __restrict__ cst,          // MODE0: cell state (4096x1024)
        const float* __restrict__ query,  // MODE0: f32 query  (4096x512)
        float* __restrict__ hout,         // MODE0: f32 h -> d_out
        bf16* __restrict__ xcat_h,        // MODE0: xcat + 512 (h slot)
        float* __restrict__ Cf,           // MODE1
        bf16* __restrict__ Cb,            // MODE2
        int ldc)
{
#if USE_ASYNC
    constexpr int SMEMB = (MODE == 0) ? 65536 : 61440;   // 3 x 20KB buffers (+Gs reuse)
#else
    constexpr int SMEMB = (MODE == 0) ? 65536 : 40960;
#endif
    __shared__ __align__(16) unsigned char smem[SMEMB];
    const int SK = 40;                  // padded LDS row stride (bf16 elems)

    const int tid  = threadIdx.x;
    const int lane = tid & 31;
    const int w    = tid >> 5;
    const int wm   = w >> 2;            // 0..1  (M 64-row half)
    const int wn   = w & 3;             // 0..3  (N 32-col quarter == gate q)
    const int half = lane >> 4;
    const int l    = lane & 15;
    const int m0   = blockIdx.x * 128;

    // global -> LDS staging: 2 threads per tile row, 32B each
    const int lr = tid >> 1;
    const int lh = tid & 1;
    int brow;
    if (MODE == 0) brow = (lr >> 5) * NH + (int)blockIdx.y * 32 + (lr & 31); // gate quadrants
    else           brow = (int)blockIdx.y * 128 + lr;
    const bf16* agp = A + (size_t)(m0 + lr) * lda + lh * 16;
    const bf16* bgp = B + (size_t)brow * ldb + lh * 16;

    v8f acc[4][2];
    #pragma unroll
    for (int mt = 0; mt < 4; ++mt)
        #pragma unroll
        for (int nt = 0; nt < 2; ++nt)
            #pragma unroll
            for (int e = 0; e < 8; ++e) acc[mt][nt][e] = 0.f;

    const int nk = K >> 5;

    // fragment consumption from one staged buffer
    auto doComp = [&](const bf16* Asb, const bf16* Bsb) {
        Frag32 fb[2];
        #pragma unroll
        for (int nt = 0; nt < 2; ++nt) {
            const bf16* pb = Bsb + (wn * 32 + nt * 16 + l) * SK + half * 16;
            fb[nt].f4[0] = ((const v4f*)pb)[0];
            fb[nt].f4[1] = ((const v4f*)pb)[1];
        }
        #pragma unroll
        for (int mt = 0; mt < 4; ++mt) {
            Frag32 fa;
            const bf16* pa = Asb + (wm * 64 + mt * 16 + l) * SK + half * 16;
            fa.f4[0] = *(const v4f*)pa;
            fa.f4[1] = *(const v4f*)(pa + 8);
            acc[mt][0] = __builtin_amdgcn_wmma_f32_16x16x32_bf16(
                false, fa.v, false, fb[0].v, (short)0, acc[mt][0], false, false);
            acc[mt][1] = __builtin_amdgcn_wmma_f32_16x16x32_bf16(
                false, fa.v, false, fb[1].v, (short)0, acc[mt][1], false, false);
        }
    };

#if USE_ASYNC
    // ---- triple-buffered async pipeline: buffer b at smem + b*20480,
    //      A tile at +0 (10240B), B tile at +10240 ----
    auto issue = [&](int kt, int buf) {
        const bf16* ag = agp + kt * 32;
        const bf16* bg = bgp + kt * 32;
        char* la = (char*)smem + buf * 20480 + (lr * SK + lh * 16) * 2;
        char* lb = la + 10240;
        __builtin_amdgcn_global_load_async_to_lds_b128(
            (__attribute__((address_space(1))) v4i_*)ag,
            (__attribute__((address_space(3))) v4i_*)la, 0, 0);
        __builtin_amdgcn_global_load_async_to_lds_b128(
            (__attribute__((address_space(1))) v4i_*)ag,
            (__attribute__((address_space(3))) v4i_*)la, 16, 0);
        __builtin_amdgcn_global_load_async_to_lds_b128(
            (__attribute__((address_space(1))) v4i_*)bg,
            (__attribute__((address_space(3))) v4i_*)lb, 0, 0);
        __builtin_amdgcn_global_load_async_to_lds_b128(
            (__attribute__((address_space(1))) v4i_*)bg,
            (__attribute__((address_space(3))) v4i_*)lb, 16, 0);
    };

    issue(0, 0);
    int cur = 0, nxt = 1;
    for (int kt = 0; kt < nk; ++kt) {
        if (kt + 1 < nk) {
            issue(kt + 1, nxt);
            wait_async<4>();          // cur's 4 loads done; nxt's 4 may fly
        } else {
            wait_async<0>();
        }
        __syncthreads();              // all waves' loads into cur complete
        const bf16* Asb = (const bf16*)((char*)smem + cur * 20480);
        doComp(Asb, Asb + 5120);
        cur = nxt; nxt = (nxt == 2) ? 0 : nxt + 1;
    }
    if (MODE == 0) __syncthreads();   // before Gs overwrites staged buffers
#else
    // ---- double-buffered VGPR staging fallback ----
    bf16* As = (bf16*)smem;             // [2][128*40]
    bf16* Bs = (bf16*)(smem + 20480);   // [2][128*40]
    const int BUFE = 128 * 40;
    {
        v4f a0 = ((const v4f*)agp)[0], a1 = ((const v4f*)agp)[1];
        v4f b0 = ((const v4f*)bgp)[0], b1 = ((const v4f*)bgp)[1];
        bf16* ad = As + lr * SK + lh * 16;
        bf16* bd = Bs + lr * SK + lh * 16;
        ((v4f*)ad)[0] = a0; ((v4f*)ad)[1] = a1;
        ((v4f*)bd)[0] = b0; ((v4f*)bd)[1] = b1;
    }
    __syncthreads();
    for (int kt = 0; kt < nk; ++kt) {
        v4f a0, a1, b0, b1;
        const bool pf = (kt + 1 < nk);
        if (pf) {
            const bf16* ap = agp + (kt + 1) * 32;
            const bf16* bp = bgp + (kt + 1) * 32;
            a0 = ((const v4f*)ap)[0]; a1 = ((const v4f*)ap)[1];
            b0 = ((const v4f*)bp)[0]; b1 = ((const v4f*)bp)[1];
        }
        doComp(As + (kt & 1) * BUFE, Bs + (kt & 1) * BUFE);
        if (pf) {
            const int nb = (kt + 1) & 1;
            bf16* ad = As + nb * BUFE + lr * SK + lh * 16;
            bf16* bd = Bs + nb * BUFE + lr * SK + lh * 16;
            ((v4f*)ad)[0] = a0; ((v4f*)ad)[1] = a1;
            ((v4f*)bd)[0] = b0; ((v4f*)bd)[1] = b1;
        }
        __syncthreads();
    }
#endif

    // ---------------- epilogues ----------------
    if constexpr (MODE == 0) {
        // exchange gate tiles via LDS: Gs[q][row 0..127][col 0..31] f32 (64KB)
        float* Gs = (float*)smem;
        const int j0 = blockIdx.y * 32;
        #pragma unroll
        for (int nt = 0; nt < 2; ++nt) {
            const int lc = nt * 16 + l;
            const float bias = bsum[wn * NH + j0 + lc];
            #pragma unroll
            for (int mt = 0; mt < 4; ++mt)
                #pragma unroll
                for (int e = 0; e < 8; ++e) {
                    const int row = wm * 64 + mt * 16 + e + half * 8;
                    Gs[(wn * 128 + row) * 32 + lc] = acc[mt][nt][e] + bias;
                }
        }
        __syncthreads();
        // fused LSTMCell: 4096 cells / 256 threads
        for (int idx = tid; idx < 128 * 32; idx += 256) {
            const int row = idx >> 5, jc = idx & 31;
            const float gi = Gs[(0 * 128 + row) * 32 + jc];
            const float gf = Gs[(1 * 128 + row) * 32 + jc];
            const float gg = Gs[(2 * 128 + row) * 32 + jc];
            const float go = Gs[(3 * 128 + row) * 32 + jc];
            const int m = m0 + row, jg = j0 + jc;
            const size_t ci = (size_t)m * NH + jg;
            const float cn = sigf(gf) * cst[ci] + sigf(gi) * tanhf(gg);
            cst[ci] = cn;
            if (jg < DIM) {
                const float hv = query[(size_t)m * DIM + jg] + sigf(go) * tanhf(cn);
                hout[(size_t)m * DIM + jg] = hv;          // final-step write is the answer
                xcat_h[(size_t)m * KCAT + jg] = (bf16)hv; // bf16 h for attention + next step
            }
        }
    } else if constexpr (MODE == 1) {
        const int n0 = blockIdx.y * 128;
        #pragma unroll
        for (int nt = 0; nt < 2; ++nt) {
            const int n = n0 + wn * 32 + nt * 16 + l;
            #pragma unroll
            for (int mt = 0; mt < 4; ++mt)
                #pragma unroll
                for (int e = 0; e < 8; ++e) {
                    const int row = m0 + wm * 64 + mt * 16 + e + half * 8;
                    Cf[(size_t)row * ldc + n] = acc[mt][nt][e];
                }
        }
    } else {
        const int n0 = blockIdx.y * 128;
        #pragma unroll
        for (int nt = 0; nt < 2; ++nt) {
            const int n = n0 + wn * 32 + nt * 16 + l;
            #pragma unroll
            for (int mt = 0; mt < 4; ++mt)
                #pragma unroll
                for (int e = 0; e < 8; ++e) {
                    const int row = m0 + wm * 64 + mt * 16 + e + half * 8;
                    Cb[(size_t)row * ldc + n] = (bf16)acc[mt][nt][e];
                }
        }
    }
}

// Row softmax over 128 logits, one wave32 per row, bf16 output.
__global__ void __launch_bounds__(256)
softmax_kernel(const float* __restrict__ logits, bf16* __restrict__ attn)
{
    const int w = threadIdx.x >> 5, lane = threadIdx.x & 31;
    const int row = blockIdx.x * 8 + w;
    const float* lp = logits + (size_t)row * NFEW;
    float v[4], mx = -3.4e38f;
    #pragma unroll
    for (int j = 0; j < 4; ++j) { v[j] = lp[lane + j * 32]; mx = fmaxf(mx, v[j]); }
    #pragma unroll
    for (int off = 16; off > 0; off >>= 1) mx = fmaxf(mx, __shfl_xor(mx, off, 32));
    float s = 0.f;
    #pragma unroll
    for (int j = 0; j < 4; ++j) { v[j] = __expf(v[j] - mx); s += v[j]; }
    #pragma unroll
    for (int off = 16; off > 0; off >>= 1) s += __shfl_xor(s, off, 32);
    const float inv = 1.f / s;
    bf16* op = attn + (size_t)row * NFEW;
    #pragma unroll
    for (int j = 0; j < 4; ++j) op[lane + j * 32] = (bf16)(v[j] * inv);
}

// One-time per launch: bf16 packing + state init (deterministic each call).
__global__ void __launch_bounds__(256)
init_kernel(const float* __restrict__ Wih, const float* __restrict__ Whh,
            const float* __restrict__ q,   const float* __restrict__ sup,
            const float* __restrict__ bih, const float* __restrict__ bhh,
            bf16* __restrict__ Wcat, bf16* __restrict__ xcat,
            bf16* __restrict__ sB,   bf16* __restrict__ sT,
            float* __restrict__ bsum, float* __restrict__ cst)
{
    const long long t = (long long)blockIdx.x * blockDim.x + threadIdx.x;
    const long long st = (long long)gridDim.x * blockDim.x;
    for (long long i = t; i < (long long)NB * KCAT; i += st) {
        const int row = (int)(i / KCAT), col = (int)(i % KCAT);
        const float wv = (col < DIM) ? Wih[(size_t)row * DIM + col]
                                     : Whh[(size_t)row * NH + (col - DIM)];
        Wcat[i] = (bf16)wv;
        xcat[i] = (col < DIM) ? (bf16)q[(size_t)row * DIM + col] : (bf16)0.f;
    }
    for (long long i = t; i < (long long)NFEW * DIM; i += st) sB[i] = (bf16)sup[i];
    for (long long i = t; i < (long long)DIM * NFEW; i += st) {
        const int d = (int)(i >> 7), f = (int)(i & (NFEW - 1));
        sT[i] = (bf16)sup[(size_t)f * DIM + d];
    }
    for (long long i = t; i < 4 * NH; i += st) bsum[i] = bih[i] + bhh[i];
    for (long long i = t; i < (long long)NB * NH; i += st) cst[i] = 0.f;
}

extern "C" void kernel_launch(void* const* d_in, const int* in_sizes, int n_in,
                              void* d_out, int out_size, void* d_ws, size_t ws_size,
                              hipStream_t stream)
{
    const float* sup = (const float*)d_in[0];
    const float* q   = (const float*)d_in[1];
    const float* Wih = (const float*)d_in[2];
    const float* Whh = (const float*)d_in[3];
    const float* bih = (const float*)d_in[4];
    const float* bhh = (const float*)d_in[5];
    float* out = (float*)d_out;   // (4096, 512) f32

    // workspace layout (~45.4 MB)
    char* ws = (char*)d_ws;
    bf16*  xcat   = (bf16*) (ws + 0);          // 4096x1536 bf16 = [q | h | r]
    bf16*  Wcat   = (bf16*) (ws + 12582912);   // 4096x1536 bf16 = [W_ih | W_hh]
    float* cst    = (float*)(ws + 25165824);   // 4096x1024 f32 cell state
    bf16*  sB     = (bf16*) (ws + 41943040);   // 128x512 bf16 support
    bf16*  sT     = (bf16*) (ws + 42074112);   // 512x128 bf16 support^T
    float* bsum   = (float*)(ws + 42205184);   // 4096 f32
    float* logits = (float*)(ws + 42221568);   // 4096x128 f32
    bf16*  attn   = (bf16*) (ws + 44318720);   // 4096x128 bf16

    init_kernel<<<2048, 256, 0, stream>>>(Wih, Whh, q, sup, bih, bhh,
                                          Wcat, xcat, sB, sT, bsum, cst);

    for (int s = 0; s < 4; ++s) {
        const int K = (s == 0) ? DIM : KCAT;   // step 0: h_r == 0, skip W_hh part
        // gates GEMM + fused LSTMCell
        gemm_nt<0><<<dim3(32, 32), 256, 0, stream>>>(
            xcat, KCAT, Wcat, KCAT, K,
            bsum, cst, q, out, xcat + DIM, nullptr, nullptr, 0);
        // logits = h @ support^T  (4096x128, K=512)
        gemm_nt<1><<<dim3(32, 1), 256, 0, stream>>>(
            xcat + DIM, KCAT, sB, DIM, DIM,
            nullptr, nullptr, nullptr, nullptr, nullptr, logits, nullptr, NFEW);
        softmax_kernel<<<NB / 8, 256, 0, stream>>>(logits, attn);
        // r = attn @ support  (4096x512, K=128) -> bf16 straight into xcat r slot
        gemm_nt<2><<<dim3(32, 4), 256, 0, stream>>>(
            attn, NFEW, sT, NFEW, NFEW,
            nullptr, nullptr, nullptr, nullptr, nullptr, nullptr, xcat + NH, KCAT);
    }
}
